// GraphSAGE_4733053960253
// MI455X (gfx1250) — compile-verified, matrix operand order checked by reading, not run on hardware
//
#include <hip/hip_runtime.h>
#include <hip/hip_bf16.h>

// ---------------------------------------------------------------------------
// GraphSAGE forward for MI455X (gfx1250, wave32, WMMA).
//   - edge scatter-add via global_atomic_add_f32 (HBM/L2 bound part)
//   - all 128x128 GEMMs via v_wmma_f32_16x16x32_bf16 (f32 accumulate)
//   - A-tiles staged through LDS as bf16: one f32->bf16 convert per element,
//     half the LDS bandwidth, no per-wave redundant cvts
//   - predictor fully fused (gather -> MLP -> 128->1 dot) with LDS reduction
// ---------------------------------------------------------------------------

typedef __attribute__((ext_vector_type(16))) __bf16 v16bf;
typedef __attribute__((ext_vector_type(8)))  __bf16 v8bf;
typedef __attribute__((ext_vector_type(2)))  __bf16 bf16x2;
typedef __attribute__((ext_vector_type(8)))  float  v8f;

#define HID 128

// K index inside a 16-bit A/B fragment (CDNA5 ISA, 16-bit A 16x32 layout):
// VGPR g (holding element pair 2g,2g+1): K = (g<4 ? 2g : 16+2(g-4)) + 8*half
__device__ __forceinline__ int kmap(int g, int half) {
    return ((g < 4) ? (2 * g) : (16 + 2 * (g - 4))) + 8 * half;
}

// Load a 16-element fragment from a bf16 row (stride 128; LDS or global).
__device__ __forceinline__ v16bf frag_bf16(const __bf16* row, int kk, int half) {
    v16bf b;
#pragma unroll
    for (int g = 0; g < 8; ++g) {
        int k = kk + kmap(g, half);            // k even -> 4B aligned
        bf16x2 p = *(const bf16x2*)(row + k);
        b[2 * g + 0] = p.x;
        b[2 * g + 1] = p.y;
    }
    return b;
}

// Convert 8 f32 -> packed v8bf (lowered to 4x v_cvt_pk_bf16_f32).
__device__ __forceinline__ v8bf cvt8(float4 a, float4 b) {
    v8bf r;
    r[0] = (__bf16)a.x; r[1] = (__bf16)a.y; r[2] = (__bf16)a.z; r[3] = (__bf16)a.w;
    r[4] = (__bf16)b.x; r[5] = (__bf16)b.y; r[6] = (__bf16)b.z; r[7] = (__bf16)b.w;
    return r;
}

// ---------------------------------------------------------------------------
// Fused SAGE GEMM:  out = act( A1 @ W1 [+ (A2*scale2) @ W2] + bias )
// A*: [rows,128] f32 row-major.  W*t: [128,128] bf16, PRE-TRANSPOSED (col,k).
// One block (8 waves) -> 16 rows x 128 cols; each wave one 16x16 WMMA tile.
// A tiles converted to bf16 and staged via LDS (global rows read once/block).
// ---------------------------------------------------------------------------
template <bool HAS2, bool RELU>
__global__ void __launch_bounds__(256)
wmma_gemm128(const float* __restrict__ A1, const __bf16* __restrict__ W1t,
             const float* __restrict__ A2, const __bf16* __restrict__ W2t,
             const float* __restrict__ scale2, const float* __restrict__ bias,
             float* __restrict__ out, int rows) {
    __shared__ __bf16 sA[16 * HID];
    __shared__ __bf16 sM[HAS2 ? 16 * HID : 8];

    const int tile = blockIdx.x;               // 16-row stripe
    const int tid  = threadIdx.x;

    {   // stage A tile (and pre-scaled mean tile) into LDS: 8 elems/thread
        const int lrow = tid >> 4;
        const int c0   = (tid & 15) * 8;
        int gr = tile * 16 + lrow;
        if (gr >= rows) gr = rows - 1;         // clamp tail
        const float* p = A1 + (size_t)gr * HID + c0;
        float4 v0 = *(const float4*)(p);
        float4 v1 = *(const float4*)(p + 4);
        *(v8bf*)(sA + lrow * HID + c0) = cvt8(v0, v1);
        if (HAS2) {
            const float s2 = scale2[gr];
            const float* q = A2 + (size_t)gr * HID + c0;
            float4 u0 = *(const float4*)(q);
            float4 u1 = *(const float4*)(q + 4);
            float4 w0 = { u0.x * s2, u0.y * s2, u0.z * s2, u0.w * s2 };
            float4 w1 = { u1.x * s2, u1.y * s2, u1.z * s2, u1.w * s2 };
            *(v8bf*)(sM + lrow * HID + c0) = cvt8(w0, w1);
        }
    }
    __syncthreads();

    const int wave = tid >> 5;                 // 0..7 -> 16-col tile
    const int lane = tid & 31;
    const int mloc = lane & 15;
    const int half = lane >> 4;
    const int col  = wave * 16 + mloc;         // B/C/D column this lane owns

    const __bf16* a1 = sA + mloc * HID;
    const __bf16* w1 = W1t + (size_t)col * HID;

    v8f acc = {};
#pragma unroll
    for (int kk = 0; kk < HID; kk += 32) {
        v16bf a = frag_bf16(a1, kk, half);
        v16bf b = frag_bf16(w1, kk, half);
        acc = __builtin_amdgcn_wmma_f32_16x16x32_bf16(false, a, false, b,
                                                      (short)0, acc, false, false);
    }
    if (HAS2) {
        const __bf16* a2 = sM + mloc * HID;
        const __bf16* w2 = W2t + (size_t)col * HID;
#pragma unroll
        for (int kk = 0; kk < HID; kk += 32) {
            v16bf a = frag_bf16(a2, kk, half);
            v16bf b = frag_bf16(w2, kk, half);
            acc = __builtin_amdgcn_wmma_f32_16x16x32_bf16(false, a, false, b,
                                                          (short)0, acc, false, false);
        }
    }

    const float bv = bias[col];
#pragma unroll
    for (int j = 0; j < 8; ++j) {              // D: vgpr j -> M = j + 8*half
        int r = tile * 16 + j + 8 * half;
        if (r < rows) {
            float v = acc[j] + bv;
            if (RELU) v = fmaxf(v, 0.0f);
            out[(size_t)r * HID + col] = v;
        }
    }
}

// ---------------------------------------------------------------------------
// Fully fused link predictor, one 16-row tile per block:
//   zin = bf16(h[src]*h[dst])        (staged in LDS)
//   z1  = bf16(relu(zin @ Wp1 + bp1))(WMMA, tile kept in LDS)
//   z2  = relu(z1 @ Wp2 + bp2)       (WMMA, in registers, f32)
//   out = z2 @ Wp3 + bp3             (shfl_xor + ds_add_f32 reduction)
// Rows [0,P) = pos pairs, [P,2P) = neg pairs. out: [2P] f32.
// ---------------------------------------------------------------------------
__global__ void __launch_bounds__(256)
predictor_fused(const float* __restrict__ h,
                const int* __restrict__ ps, const int* __restrict__ pd,
                const int* __restrict__ ns, const int* __restrict__ nd,
                const __bf16* __restrict__ W1t, const float* __restrict__ bv1,
                const __bf16* __restrict__ W2t, const float* __restrict__ bv2,
                const float* __restrict__ Wp3, const float* __restrict__ bp3,
                float* __restrict__ out, int P) {
    __shared__ __bf16 zin[16 * HID];
    __shared__ __bf16 z1[16 * HID];
    __shared__ float  red[16];

    const int tile = blockIdx.x;
    const int tid  = threadIdx.x;

    {   // stage gathered edge features: 8 elems/thread
        const int lrow = tid >> 4;
        const int c0   = (tid & 15) * 8;
        const int r    = tile * 16 + lrow;     // 2P is a multiple of 16
        int s, d;
        if (r < P) { s = ps[r];     d = pd[r];     }
        else       { s = ns[r - P]; d = nd[r - P]; }
        const float* hs = h + (size_t)s * HID + c0;
        const float* hd = h + (size_t)d * HID + c0;
        float4 a0 = *(const float4*)(hs);
        float4 a1 = *(const float4*)(hs + 4);
        float4 b0 = *(const float4*)(hd);
        float4 b1 = *(const float4*)(hd + 4);
        float4 o0 = { a0.x * b0.x, a0.y * b0.y, a0.z * b0.z, a0.w * b0.w };
        float4 o1 = { a1.x * b1.x, a1.y * b1.y, a1.z * b1.z, a1.w * b1.w };
        *(v8bf*)(zin + lrow * HID + c0) = cvt8(o0, o1);
    }
    __syncthreads();

    const int wave = tid >> 5;
    const int lane = tid & 31;
    const int mloc = lane & 15;
    const int half = lane >> 4;
    const int col  = wave * 16 + mloc;

    // ---- layer p1: z1 = relu(zin @ Wp1 + bp1) ----
    v8f acc = {};
    {
        const __bf16* ar = zin + mloc * HID;
        const __bf16* wr = W1t + (size_t)col * HID;
#pragma unroll
        for (int kk = 0; kk < HID; kk += 32) {
            v16bf a = frag_bf16(ar, kk, half);
            v16bf b = frag_bf16(wr, kk, half);
            acc = __builtin_amdgcn_wmma_f32_16x16x32_bf16(false, a, false, b,
                                                          (short)0, acc, false, false);
        }
        const float bb = bv1[col];
#pragma unroll
        for (int j = 0; j < 8; ++j)
            z1[(j + 8 * half) * HID + col] = (__bf16)fmaxf(acc[j] + bb, 0.0f);
    }
    if (tid < 16) red[tid] = 0.0f;
    __syncthreads();

    // ---- layer p2: z2 = relu(z1 @ Wp2 + bp2) ----
    acc = (v8f){};
    {
        const __bf16* ar = z1 + mloc * HID;
        const __bf16* wr = W2t + (size_t)col * HID;
#pragma unroll
        for (int kk = 0; kk < HID; kk += 32) {
            v16bf a = frag_bf16(ar, kk, half);
            v16bf b = frag_bf16(wr, kk, half);
            acc = __builtin_amdgcn_wmma_f32_16x16x32_bf16(false, a, false, b,
                                                          (short)0, acc, false, false);
        }
    }

    // ---- fused z2 @ Wp3 + bp3 ----
    const float w3  = Wp3[col];
    const float bb2 = bv2[col];
#pragma unroll
    for (int j = 0; j < 8; ++j) {
        float v = fmaxf(acc[j] + bb2, 0.0f) * w3;
#pragma unroll
        for (int off = 1; off < 16; off <<= 1)   // reduce across the 16 N-lanes
            v += __shfl_xor(v, off, 32);
        if (mloc == 0) atomicAdd(&red[j + 8 * half], v);  // cross-wave via LDS
    }
    __syncthreads();
    if (tid < 16)
        out[tile * 16 + tid] = red[tid] + bp3[0];
}

// ---------------------------------------------------------------------------
// Support kernels
// ---------------------------------------------------------------------------
__global__ void zero_f32(float* __restrict__ p, size_t n) {
    size_t i = (size_t)blockIdx.x * blockDim.x + threadIdx.x;
    size_t stride = (size_t)gridDim.x * blockDim.x;
    for (; i < n; i += stride) p[i] = 0.0f;
}

__global__ void deg_count(const int* __restrict__ dst, float* __restrict__ deg, int E) {
    int e = blockIdx.x * blockDim.x + threadIdx.x;
    if (e < E) atomicAdd(&deg[dst[e]], 1.0f);
}

__global__ void inv_deg_k(float* __restrict__ d, int N) {
    int i = blockIdx.x * blockDim.x + threadIdx.x;
    if (i < N) d[i] = 1.0f / fmaxf(d[i], 1.0f);
}

// W: [128,128] f32 (k,col) -> Wt: [128,128] bf16 (col,k)
__global__ void cvt_wt(const float* __restrict__ W, __bf16* __restrict__ Wt) {
    int t = blockIdx.x * blockDim.x + threadIdx.x;   // 16384 threads
    int col = t >> 7, k = t & 127;
    Wt[t] = (__bf16)W[k * HID + col];
}

// agg[dst] += h[src]  (4 floats per thread, E*32 threads)
__global__ void scatter_add(const float* __restrict__ h, const int* __restrict__ src,
                            const int* __restrict__ dst, float* __restrict__ agg, int E) {
    long long t = (long long)blockIdx.x * blockDim.x + threadIdx.x;
    if (t >= (long long)E * 32) return;
    int e = (int)(t >> 5);
    int c = ((int)t & 31) * 4;
    int s = src[e], d = dst[e];
    float4 v = *(const float4*)(h + (size_t)s * HID + c);
    float* a = agg + (size_t)d * HID + c;
    atomicAdd(a + 0, v.x);
    atomicAdd(a + 1, v.y);
    atomicAdd(a + 2, v.z);
    atomicAdd(a + 3, v.w);
}

// ---------------------------------------------------------------------------
extern "C" void kernel_launch(void* const* d_in, const int* in_sizes, int n_in,
                              void* d_out, int out_size, void* d_ws, size_t ws_size,
                              hipStream_t stream) {
    (void)n_in; (void)out_size; (void)ws_size;

    const float* x       = (const float*)d_in[0];
    const int* edge_src  = (const int*)d_in[1];
    const int* edge_dst  = (const int*)d_in[2];
    const int* pos_src   = (const int*)d_in[3];
    const int* pos_dst   = (const int*)d_in[4];
    const int* neg_src   = (const int*)d_in[5];
    const int* neg_dst   = (const int*)d_in[6];
    const float* W_self0 = (const float*)d_in[7];
    const float* W_neigh0= (const float*)d_in[8];
    const float* b0      = (const float*)d_in[9];
    const float* W_self1 = (const float*)d_in[10];
    const float* W_neigh1= (const float*)d_in[11];
    const float* b1      = (const float*)d_in[12];
    const float* W_self2 = (const float*)d_in[13];
    const float* W_neigh2= (const float*)d_in[14];
    const float* b2      = (const float*)d_in[15];
    const float* Wp1     = (const float*)d_in[16];
    const float* bp1     = (const float*)d_in[17];
    const float* Wp2     = (const float*)d_in[18];
    const float* bp2     = (const float*)d_in[19];
    const float* Wp3     = (const float*)d_in[20];
    const float* bp3     = (const float*)d_in[21];

    const int N = in_sizes[0] / HID;   // 50000
    const int E = in_sizes[1];         // 640000
    const int P = in_sizes[3];         // 100000
    const int R = 2 * P;               // predictor rows (pos then neg)

    // --- carve workspace (256B aligned slices) ---
    char* w = (char*)d_ws;
    auto carve = [&](size_t bytes) {
        void* p = (void*)w;
        w += (bytes + 255) & ~(size_t)255;
        return p;
    };
    float*  invdeg = (float*)carve((size_t)N * 4);
    float*  agg    = (float*)carve((size_t)N * HID * 4);
    float*  hA     = (float*)carve((size_t)N * HID * 4);
    float*  hB     = (float*)carve((size_t)N * HID * 4);
    __bf16* Wt[8];
    for (int i = 0; i < 8; ++i) Wt[i] = (__bf16*)carve((size_t)HID * HID * 2);

    const int B256 = 256;
    const int zgrid = 2048;
    const int gemmN = (N + 15) / 16;   // 3125
    const int predR = (R + 15) / 16;   // 12500
    const long long scT = (long long)E * 32;
    const int scGrid = (int)((scT + B256 - 1) / B256);

    // degree -> inv_deg
    zero_f32<<<zgrid, B256, 0, stream>>>(invdeg, (size_t)N);
    deg_count<<<(E + B256 - 1) / B256, B256, 0, stream>>>(edge_dst, invdeg, E);
    inv_deg_k<<<(N + B256 - 1) / B256, B256, 0, stream>>>(invdeg, N);

    // bf16-transposed weights (once per launch)
    const float* Wsrc[8] = { W_self0, W_neigh0, W_self1, W_neigh1,
                             W_self2, W_neigh2, Wp1, Wp2 };
    for (int i = 0; i < 8; ++i)
        cvt_wt<<<(HID * HID) / B256, B256, 0, stream>>>(Wsrc[i], Wt[i]);

    // ---- layer 0: x -> hA (relu) ----
    zero_f32<<<zgrid, B256, 0, stream>>>(agg, (size_t)N * HID);
    scatter_add<<<scGrid, B256, 0, stream>>>(x, edge_src, edge_dst, agg, E);
    wmma_gemm128<true, true><<<gemmN, B256, 0, stream>>>(
        x, Wt[0], agg, Wt[1], invdeg, b0, hA, N);

    // ---- layer 1: hA -> hB (relu) ----
    zero_f32<<<zgrid, B256, 0, stream>>>(agg, (size_t)N * HID);
    scatter_add<<<scGrid, B256, 0, stream>>>(hA, edge_src, edge_dst, agg, E);
    wmma_gemm128<true, true><<<gemmN, B256, 0, stream>>>(
        hA, Wt[2], agg, Wt[3], invdeg, b1, hB, N);

    // ---- layer 2: hB -> hA (no relu) ----
    zero_f32<<<zgrid, B256, 0, stream>>>(agg, (size_t)N * HID);
    scatter_add<<<scGrid, B256, 0, stream>>>(hB, edge_src, edge_dst, agg, E);
    wmma_gemm128<true, false><<<gemmN, B256, 0, stream>>>(
        hB, Wt[4], agg, Wt[5], invdeg, b2, hA, N);

    // ---- fused predictor: gather -> MLP(2x WMMA) -> 128->1 dot ----
    predictor_fused<<<predR, B256, 0, stream>>>(
        hA, pos_src, pos_dst, neg_src, neg_dst,
        Wt[6], bp1, Wt[7], bp2, Wp3, bp3, (float*)d_out, P);
}
